// SingleInstrumentPredictorRNN_5299989643602
// MI455X (gfx1250) — matrix operand order repaired and checked by reading, not compiled, not run on hardware
//
#include <hip/hip_runtime.h>

typedef __bf16 bf16;
typedef __attribute__((ext_vector_type(16))) __bf16 v16bf;
typedef __attribute__((ext_vector_type(8)))  __bf16 v8bf;
typedef __attribute__((ext_vector_type(8)))  float  v8f;

#define BB  8192
#define TN  658
#define TP  672     // T padded to multiple of 32 (K of layer 1)
#define II  7
#define D1  1024
#define D2  512
#define D3  128
#define NP4 768     // layer-4 N (=T) padded to multiple of 128

// ---------------------------------------------------------------------------
// Kernel 1: input projection xw[b,t] = x[b,t,:] . W_ih + b_ih
// One thread per (b,t): lanes read consecutive 28B chunks -> coalesced stream.
// ---------------------------------------------------------------------------
__global__ __launch_bounds__(256) void xw_kernel(
    const float* __restrict__ x, const float* __restrict__ W_ih,
    const float* __restrict__ b_ih, float* __restrict__ xw)
{
  long long idx = (long long)blockIdx.x * 256 + threadIdx.x;
  if (idx >= (long long)BB * TN) return;
  const float* xp = x + idx * II;
  float acc = b_ih[0];
#pragma unroll
  for (int i = 0; i < II; ++i) acc += xp[i] * W_ih[i];
  xw[idx] = acc;
}

// ---------------------------------------------------------------------------
// Kernel 2: scalar RNN recurrence (H == 1). One thread per batch row.
// Chain per step is just tanh(fma) — minimal serial latency. Reads/writes
// stream contiguously per thread, so HBM lines are fully consumed.
// Writes hs as bf16 into actb[B, TP] (zero padded), h_last as f32.
// ---------------------------------------------------------------------------
__global__ __launch_bounds__(256) void rnn_kernel(
    const float* __restrict__ xw, const float* __restrict__ h0,
    const float* __restrict__ W_hh, const float* __restrict__ b_hh,
    bf16* __restrict__ actb, float* __restrict__ hlast)
{
  int b = blockIdx.x * 256 + threadIdx.x;
  if (b >= BB) return;
  const float whh = W_hh[0], bh = b_hh[0];
  float h = h0[b];
  const float* xp = xw + (long long)b * TN;
  bf16* ap = actb + (long long)b * TP;
  for (int t = 0; t < TN; ++t) {
    h = tanhf(fmaf(h, whh, xp[t] + bh));
    ap[t] = (bf16)h;
  }
#pragma unroll
  for (int t = TN; t < TP; ++t) ap[t] = (bf16)0.0f;
  hlast[b] = h;
}

// ---------------------------------------------------------------------------
// Kernel 3: f32 -> bf16 weight convert with zero padding to (Rp, Cp).
// ---------------------------------------------------------------------------
__global__ __launch_bounds__(256) void pad_convert(
    const float* __restrict__ src, bf16* __restrict__ dst,
    int R, int C, int Rp, int Cp)
{
  int idx = blockIdx.x * 256 + threadIdx.x;
  if (idx >= Rp * Cp) return;
  int r = idx / Cp, c = idx % Cp;
  float v = (r < R && c < C) ? src[r * C + c] : 0.0f;
  dst[idx] = (bf16)v;
}

// ---------------------------------------------------------------------------
// WMMA fragment loaders per CDNA5 ISA VGPR layouts (wave32).
// A 16x32 bf16: lane(&15)=M, kh=lane>>4; elems 0..7 = K[kh*8..+7],
//               elems 8..15 = K[16+kh*8..+7]  (two 16B contiguous chunks)
// B 32x16 bf16: lane(&15)=N, kh=lane>>4; elems 0..7 = K[kh*8..+7],
//               elems 8..15 = K[16+kh*8..+7]  (W stored row-major over K)
// ---------------------------------------------------------------------------
__device__ __forceinline__ v16bf ld_a_frag(const bf16* p) {
  v8bf lo = *(const v8bf*)(p);
  v8bf hi = *(const v8bf*)(p + 16);
  v16bf r;
#pragma unroll
  for (int i = 0; i < 8; ++i) { r[i] = lo[i]; r[i + 8] = hi[i]; }
  return r;
}
__device__ __forceinline__ v16bf ld_b_frag(const bf16* p) {
  v8bf lo = *(const v8bf*)(p);
  v8bf hi = *(const v8bf*)(p + 16);
  v16bf r;
#pragma unroll
  for (int i = 0; i < 8; ++i) { r[i] = lo[i]; r[i + 8] = hi[i]; }
  return r;
}

// ---------------------------------------------------------------------------
// GEMM: out[M,N] = act( A[M,Kp](bf16) @ W[N,Kp]^T(bf16) + bias ), f32 accum.
// Block = 256 threads = 8 waves -> 128(M) x 128(N) tile.
// Wave = 32(M) x 64(N) = 2x4 v_wmma_f32_16x16x32_bf16 tiles.
// Weights/activations are L2-resident (192MB), so direct global fragment
// loads are fine; no LDS staging needed at this arithmetic intensity.
// ---------------------------------------------------------------------------
template<bool RELU, bool OUT_F32>
__global__ __launch_bounds__(256) void gemm_bf16_wmma(
    const bf16* __restrict__ A, const bf16* __restrict__ W,
    const float* __restrict__ bias,
    bf16* __restrict__ outb, float* __restrict__ outf,
    int Kp, int ldo, int Nvalid)
{
  const int lane = threadIdx.x & 31;
  const int wv   = threadIdx.x >> 5;
  const int wm   = wv & 3;            // 4 waves over M
  const int wn   = wv >> 2;           // 2 waves over N
  const int m0   = blockIdx.y * 128 + wm * 32;
  const int n0   = blockIdx.x * 128 + wn * 64;
  const int lr   = lane & 15;
  const int kh   = lane >> 4;

  v8f z = {};
  v8f acc[2][4];
#pragma unroll
  for (int i = 0; i < 2; ++i)
#pragma unroll
    for (int j = 0; j < 4; ++j) acc[i][j] = z;

  const bf16* arow[2];
#pragma unroll
  for (int i = 0; i < 2; ++i)
    arow[i] = A + (long long)(m0 + i * 16 + lr) * Kp + kh * 8;
  const bf16* brow[4];
#pragma unroll
  for (int j = 0; j < 4; ++j)
    brow[j] = W + (long long)(n0 + j * 16 + lr) * Kp + kh * 8;

  for (int k0 = 0; k0 < Kp; k0 += 32) {
    v16bf af[2], bfr[4];
#pragma unroll
    for (int i = 0; i < 2; ++i) af[i] = ld_a_frag(arow[i] + k0);
#pragma unroll
    for (int j = 0; j < 4; ++j) bfr[j] = ld_b_frag(brow[j] + k0);
#pragma unroll
    for (int i = 0; i < 2; ++i)
#pragma unroll
      for (int j = 0; j < 4; ++j)
        acc[i][j] = __builtin_amdgcn_wmma_f32_16x16x32_bf16(
            false, af[i], false, bfr[j], (short)0, acc[i][j], false, false);
  }

  // Epilogue. C/D layout: VGPR r -> M = 8*(lane>=16)+r, N = lane&15.
#pragma unroll
  for (int i = 0; i < 2; ++i) {
#pragma unroll
    for (int j = 0; j < 4; ++j) {
      int col = n0 + j * 16 + lr;
      float bv = (col < Nvalid) ? bias[col] : 0.0f;
#pragma unroll
      for (int r = 0; r < 8; ++r) {
        int row = m0 + i * 16 + kh * 8 + r;
        float v = acc[i][j][r] + bv;
        if (RELU) v = fmaxf(v, 0.0f);
        if (OUT_F32) {
          if (col < Nvalid) outf[(long long)row * ldo + col] = v;
        } else {
          outb[(long long)row * ldo + col] = (bf16)v;
        }
      }
    }
  }
}

// ---------------------------------------------------------------------------
extern "C" void kernel_launch(void* const* d_in, const int* in_sizes, int n_in,
                              void* d_out, int out_size, void* d_ws, size_t ws_size,
                              hipStream_t stream) {
  const float* x    = (const float*)d_in[0];
  const float* h0   = (const float*)d_in[1];
  const float* W_ih = (const float*)d_in[2];
  const float* W_hh = (const float*)d_in[3];
  const float* b_ih = (const float*)d_in[4];
  const float* b_hh = (const float*)d_in[5];
  const float* w1   = (const float*)d_in[6];
  const float* b1   = (const float*)d_in[7];
  const float* w2   = (const float*)d_in[8];
  const float* b2   = (const float*)d_in[9];
  const float* w3   = (const float*)d_in[10];
  const float* b3   = (const float*)d_in[11];
  const float* w4   = (const float*)d_in[12];
  const float* b4   = (const float*)d_in[13];

  float* out   = (float*)d_out;                 // [B, T] f32
  float* hlast = out + (size_t)BB * TN;         // [1, B, 1] f32

  // Workspace carve-up (256B aligned). xw and a1 have disjoint lifetimes
  // (xw is dead after rnn_kernel, a1 first written by GEMM1) -> alias them.
  char* ws = (char*)d_ws;
  size_t off = 0;
  auto alloc = [&](size_t bytes) -> void* {
    void* p = ws + off;
    off += (bytes + 255) & ~(size_t)255;
    return p;
  };
  size_t xw_bytes = (size_t)BB * TN * sizeof(float);   // 21.6 MB
  size_t a1_bytes = (size_t)BB * D1 * sizeof(bf16);    // 16.8 MB
  void* u0   = alloc(xw_bytes > a1_bytes ? xw_bytes : a1_bytes);
  float* xw  = (float*)u0;
  bf16*  a1  = (bf16*)u0;
  bf16* actb = (bf16*)alloc((size_t)BB * TP * sizeof(bf16));
  bf16* a2   = (bf16*)alloc((size_t)BB * D2 * sizeof(bf16));
  bf16* a3   = (bf16*)alloc((size_t)BB * D3 * sizeof(bf16));
  bf16* w1b  = (bf16*)alloc((size_t)D1 * TP * sizeof(bf16));
  bf16* w2b  = (bf16*)alloc((size_t)D2 * D1 * sizeof(bf16));
  bf16* w3b  = (bf16*)alloc((size_t)D3 * D2 * sizeof(bf16));
  bf16* w4b  = (bf16*)alloc((size_t)NP4 * D3 * sizeof(bf16));

  // Weight conversions (independent of xw/rnn; same stream is fine).
  pad_convert<<<(D1 * TP + 255) / 256, 256, 0, stream>>>(w1, w1b, D1, TN, D1, TP);
  pad_convert<<<(D2 * D1 + 255) / 256, 256, 0, stream>>>(w2, w2b, D2, D1, D2, D1);
  pad_convert<<<(D3 * D2 + 255) / 256, 256, 0, stream>>>(w3, w3b, D3, D2, D3, D2);
  pad_convert<<<(NP4 * D3 + 255) / 256, 256, 0, stream>>>(w4, w4b, TN, D3, NP4, D3);

  // Input projection (bandwidth phase) then serial recurrence (latency phase).
  long long npair = (long long)BB * TN;
  xw_kernel<<<(unsigned)((npair + 255) / 256), 256, 0, stream>>>(x, W_ih, b_ih, xw);
  rnn_kernel<<<BB / 256, 256, 0, stream>>>(xw, h0, W_hh, b_hh, actb, hlast);

  // MLP via bf16 WMMA, f32 accumulate.
  gemm_bf16_wmma<true,  false><<<dim3(D1 / 128, BB / 128), 256, 0, stream>>>(
      actb, w1b, b1, a1, nullptr, TP, D1, D1);
  gemm_bf16_wmma<true,  false><<<dim3(D2 / 128, BB / 128), 256, 0, stream>>>(
      a1, w2b, b2, a2, nullptr, D1, D2, D2);
  gemm_bf16_wmma<true,  false><<<dim3(D3 / 128, BB / 128), 256, 0, stream>>>(
      a2, w3b, b3, a3, nullptr, D2, D3, D3);
  gemm_bf16_wmma<false, true ><<<dim3(NP4 / 128, BB / 128), 256, 0, stream>>>(
      a3, w4b, b4, nullptr, out, D3, TN, TN);
}
